// Phase2_TransformerDecoderLayer_24489903522109
// MI455X (gfx1250) — matrix operand order repaired and checked by reading, not compiled, not run on hardware
//
#include <hip/hip_runtime.h>
#include <hip/hip_bf16.h>

// ---------------------------------------------------------------------------
// CDNA5 / gfx1250 decode-step transformer decoder layer.
// B=32, S=2048, E=256, P=512, D=1024, H=16, HD=64, DFF=4096, Tq=1.
//
// T=1 lets us fold K-projection into the query (u = Wk_h @ q_h / 8) and pull
// the V-projection outside the attention sum. The layer becomes HBM-bound:
// two streaming passes over the memory banks (~740MB -> ~32us @ 23.3TB/s).
// Dense math uses v_wmma_f32_16x16x32_bf16 (wave32 WMMA); probability chunks
// are staged into LDS with global_load_async_to_lds_b128 (ASYNCcnt).
// ---------------------------------------------------------------------------

typedef __attribute__((ext_vector_type(16))) __bf16 v16bf;
typedef __attribute__((ext_vector_type(8)))  __bf16 v8bf;
typedef __attribute__((ext_vector_type(8)))  float  v8f;

#define DMODEL 1024
#define NHEAD  16
#define HDIM   64
#define NBATCH 32

// --- WMMA fragment loaders (16x16x32 bf16, contiguous k order: k = 16*(L>>4) + element) --

// A matrix 16x32 (MxK), fp32 in memory, converted to bf16 on load.
// lane L: M = L&15 ; elements 0..15 = K[k0 + klo .. +15]
// where klo = (L>=16) ? 16 : 0.
__device__ inline v16bf load_a_frag_f32(const float* __restrict__ A, int ld,
                                        int rowbase, int k0, int lane) {
  int m   = lane & 15;
  int klo = (lane >> 4) * 16;
  const float* p = A + (size_t)(rowbase + m) * ld + k0 + klo;
  v16bf a;
#pragma unroll
  for (int i = 0; i < 8; ++i) a[i] = (__bf16)p[i];
#pragma unroll
  for (int i = 0; i < 8; ++i) a[8 + i] = (__bf16)p[8 + i];
  return a;
}

// B matrix 32x16 (KxN), fp32 row-major [K,N] in global memory.
// lane L: N = ncol + (L&15); K range = (L>=16 ? 16..31 : 0..15), element j = K[kb+j].
__device__ inline v16bf load_b_frag_f32(const float* __restrict__ W, int ldw,
                                        int k0, int ncol, int lane) {
  int n  = ncol + (lane & 15);
  int kb = (lane >> 4) * 16;
  const float* p = W + (size_t)(k0 + kb) * ldw + n;
  v16bf b;
#pragma unroll
  for (int j = 0; j < 16; ++j) b[j] = (__bf16)p[(size_t)j * ldw];
  return b;
}

// B fragment from LDS, layout ut[16 cols][1024 k] bf16 (contiguous in k).
// 32 contiguous bytes per lane, 32B-aligned -> ds_load_b128 x2.
__device__ inline v16bf load_b_frag_lds(const __bf16* ut, int k0, int lane) {
  int col = lane & 15;
  int kb  = (lane >> 4) * 16;
  const __bf16* p = ut + col * 1024 + k0 + kb;
  v8bf lo = *(const v8bf*)p;
  v8bf hi = *(const v8bf*)(p + 8);
  return __builtin_shufflevector(lo, hi, 0, 1, 2, 3, 4, 5, 6, 7,
                                 8, 9, 10, 11, 12, 13, 14, 15);
}

__device__ inline v8f wmma_bf16(v16bf a, v16bf b, v8f c) {
  return __builtin_amdgcn_wmma_f32_16x16x32_bf16(false, a, false, b,
                                                 (short)0, c, false, false);
}

// ---------------------------------------------------------------------------
// gemm32: C[32,N] = act(A[32,K] @ W[K,N] + bias) (+ resid). WMMA bf16.
// block = 256 threads (8 waves), each wave owns a 16-column tile, grid.x = N/128.
// ---------------------------------------------------------------------------
__global__ void gemm32_kernel(const float* __restrict__ A,
                              const float* __restrict__ W,
                              const float* __restrict__ bias,
                              const float* __restrict__ resid,
                              float* __restrict__ C,
                              int K, int N, int do_relu) {
  int lane = threadIdx.x & 31;
  int wave = threadIdx.x >> 5;
  int ncol = blockIdx.x * 128 + wave * 16;
  v8f c0 = {}; v8f c1 = {};
#pragma unroll 2
  for (int k0 = 0; k0 < K; k0 += 32) {
    v16bf bf = load_b_frag_f32(W, N, k0, ncol, lane);
    v16bf a0 = load_a_frag_f32(A, K, 0,  k0, lane);
    v16bf a1 = load_a_frag_f32(A, K, 16, k0, lane);
    c0 = wmma_bf16(a0, bf, c0);
    c1 = wmma_bf16(a1, bf, c1);
  }
  int n  = ncol + (lane & 15);
  int r0 = (lane >> 4) * 8;          // C layout: lanes 0-15 -> M=j, 16-31 -> M=8+j
  float bv = bias ? bias[n] : 0.0f;
#pragma unroll
  for (int j = 0; j < 8; ++j) {
    int row = r0 + j;
    float v0 = c0[j] + bv;
    float v1 = c1[j] + bv;
    if (do_relu) { v0 = fmaxf(v0, 0.0f); v1 = fmaxf(v1, 0.0f); }
    if (resid)   { v0 += resid[row * N + n]; v1 += resid[(16 + row) * N + n]; }
    C[row * N + n]        = v0;
    C[(16 + row) * N + n] = v1;
  }
}

// ---------------------------------------------------------------------------
// attn_score: scores[b,h,r] = mem[b,r,:] . u[b,h,:]   (K-projection folded)
// GEMM [S,1024] x [1024,16] per batch. u staged in LDS as bf16 (32KB).
// block = 256 threads (8 waves x 16 rows = 128 key rows). grid = (S/128, B).
// k-loop unrolled x2: two WMMAs per iteration chained on one accumulator.
// ---------------------------------------------------------------------------
__global__ void attn_score_kernel(const float* __restrict__ mem,   // [B,S,1024]
                                  const float* __restrict__ u,     // [B,16,1024]
                                  float* __restrict__ scores,      // [B,16,S]
                                  int S) {
  __shared__ __bf16 ut[NHEAD * 1024];
  int b = blockIdx.y;
  const float* ug = u + (size_t)b * NHEAD * 1024;
  for (int i = threadIdx.x; i < NHEAD * 1024; i += 256) ut[i] = (__bf16)ug[i];
  __syncthreads();

  int lane    = threadIdx.x & 31;
  int wave    = threadIdx.x >> 5;
  int rowbase = blockIdx.x * 128 + wave * 16;
  const float* memb = mem + (size_t)b * S * DMODEL;
  const float* myrow = memb + (size_t)(rowbase + (lane & 15)) * DMODEL;
  v8f c = {};
  for (int k0 = 0; k0 < DMODEL; k0 += 64) {
    __builtin_prefetch(myrow + ((k0 + 128) & (DMODEL - 1)), 0, 0);  // global_prefetch
    v16bf a0 = load_a_frag_f32(memb, DMODEL, rowbase, k0, lane);
    v16bf b0 = load_b_frag_lds(ut, k0, lane);
    c = wmma_bf16(a0, b0, c);
    v16bf a1 = load_a_frag_f32(memb, DMODEL, rowbase, k0 + 32, lane);
    v16bf b1 = load_b_frag_lds(ut, k0 + 32, lane);
    c = wmma_bf16(a1, b1, c);
  }
  int h  = lane & 15;
  int r0 = rowbase + (lane >> 4) * 8;
  float* sb = scores + ((size_t)b * NHEAD + h) * S;
#pragma unroll
  for (int j = 0; j < 8; ++j) sb[r0 + j] = c[j];
}

// ---------------------------------------------------------------------------
// softmax over keys, with folded K-bias term, pad mask, optional entity-score
// reweight + renormalize (graph attention). grid = (H, B), block = 256.
// ---------------------------------------------------------------------------
__global__ void softmax_kernel(float* __restrict__ scores,          // [B,16,S]
                               const float* __restrict__ cbias,     // [B,16]
                               const unsigned char* __restrict__ mask,  // [B,S]|null
                               const float* __restrict__ escore,    // [B,S]|null
                               int S) {
  int b = blockIdx.y, h = blockIdx.x;
  float* s = scores + ((size_t)b * NHEAD + h) * S;
  const unsigned char* mb = mask ? mask + (size_t)b * S : nullptr;
  const float* eb = escore ? escore + (size_t)b * S : nullptr;
  float cb = cbias[b * NHEAD + h];
  __shared__ float red[256];

  float mx = -3.4e38f;
  for (int r = threadIdx.x; r < S; r += 256) {
    float v = s[r] + cb;
    if (mb && mb[r]) v = -1e18f;
    mx = fmaxf(mx, v);
  }
  red[threadIdx.x] = mx; __syncthreads();
  for (int off = 128; off > 0; off >>= 1) {
    if (threadIdx.x < off) red[threadIdx.x] = fmaxf(red[threadIdx.x], red[threadIdx.x + off]);
    __syncthreads();
  }
  mx = red[0]; __syncthreads();

  float sum = 0.0f;
  for (int r = threadIdx.x; r < S; r += 256) {
    float v = s[r] + cb;
    if (mb && mb[r]) v = -1e18f;
    float e = __expf(v - mx);
    if (eb) e *= eb[r];
    s[r] = e;
    sum += e;
  }
  red[threadIdx.x] = sum; __syncthreads();
  for (int off = 128; off > 0; off >>= 1) {
    if (threadIdx.x < off) red[threadIdx.x] += red[threadIdx.x + off];
    __syncthreads();
  }
  float inv = 1.0f / fmaxf(red[0], 1e-9f);
  for (int r = threadIdx.x; r < S; r += 256) s[r] *= inv;
}

// ---------------------------------------------------------------------------
// attn_wsum: mbar[b,h,:] = sum_r probs[b,h,r] * mem[b,r,:]
// Streaming, coalesced in d. Probability chunk [16h x 64r] staged into LDS
// with global_load_async_to_lds_b128 (ASYNCcnt path), fenced by
// s_wait_asynccnt 0 + workgroup barrier. grid = (D/256, B), block = 256.
// ---------------------------------------------------------------------------
__global__ void attn_wsum_kernel(const float* __restrict__ mem,    // [B,S,1024]
                                 const float* __restrict__ probs,  // [B,16,S]
                                 float* __restrict__ mbar,         // [B,16,1024]
                                 int S) {
  __shared__ float sp[NHEAD * 64];
  int b = blockIdx.y;
  int d = blockIdx.x * 256 + threadIdx.x;
  const float* memb = mem + (size_t)b * S * DMODEL;
  const float* pb   = probs + (size_t)b * NHEAD * S;
  float acc[NHEAD];
#pragma unroll
  for (int h = 0; h < NHEAD; ++h) acc[h] = 0.0f;

  // this thread's async-copy slot: 4 consecutive probs of one head
  int sh   = threadIdx.x >> 4;          // head 0..15
  int srr  = (threadIdx.x & 15) * 4;    // row offset within chunk, 16B aligned
  unsigned lds_off = (unsigned)(uintptr_t)(&sp[sh * 64 + srr]);

  for (int r0 = 0; r0 < S; r0 += 64) {
    __syncthreads();   // previous chunk fully consumed before overwrite
    {
      unsigned long long gaddr =
          (unsigned long long)(uintptr_t)(pb + (size_t)sh * S + r0 + srr);
      asm volatile("global_load_async_to_lds_b128 %0, %1, off"
                   :: "v"(lds_off), "v"(gaddr) : "memory");
      asm volatile("s_wait_asynccnt 0x0" ::: "memory");
    }
    __syncthreads();   // all waves' async data visible in LDS
    for (int rr = 0; rr < 64; ++rr) {
      float v = memb[(size_t)(r0 + rr) * DMODEL + d];
#pragma unroll
      for (int h = 0; h < NHEAD; ++h) acc[h] = fmaf(sp[h * 64 + rr], v, acc[h]);
    }
  }
  float* ob = mbar + (size_t)b * NHEAD * DMODEL;
#pragma unroll
  for (int h = 0; h < NHEAD; ++h) ob[h * DMODEL + d] = acc[h];
}

// ---------------------------------------------------------------------------
// attn_u: u[b,h,k] = (1/8) * sum_j Wk[k, h*64+j] * q[b, h*64+j]
//         cbias[b,h] = (1/8) * sum_j bk[h*64+j] * q[b, h*64+j]
// grid = (H, B), block = 256.
// ---------------------------------------------------------------------------
__global__ void attn_u_kernel(const float* __restrict__ q,   // [B,1024]
                              const float* __restrict__ Wk,  // [1024,1024]
                              const float* __restrict__ bk,  // [1024]
                              float* __restrict__ u,         // [B,16,1024]
                              float* __restrict__ cbias) {   // [B,16]
  int h = blockIdx.x, b = blockIdx.y;
  __shared__ float qh[HDIM];
  __shared__ float red[HDIM];
  int c = h * HDIM;
  if (threadIdx.x < HDIM) qh[threadIdx.x] = 0.125f * q[b * DMODEL + c + threadIdx.x];
  __syncthreads();
#pragma unroll
  for (int kk = 0; kk < 4; ++kk) {
    int k = threadIdx.x + kk * 256;
    const float* wrow = Wk + (size_t)k * DMODEL + c;
    float acc = 0.0f;
#pragma unroll
    for (int j = 0; j < HDIM; ++j) acc = fmaf(wrow[j], qh[j], acc);
    u[((size_t)b * NHEAD + h) * DMODEL + k] = acc;
  }
  if (threadIdx.x < HDIM) red[threadIdx.x] = bk[c + threadIdx.x] * qh[threadIdx.x];
  __syncthreads();
  if (threadIdx.x == 0) {
    float s = 0.0f;
    for (int i = 0; i < HDIM; ++i) s += red[i];
    cbias[b * NHEAD + h] = s;
  }
}

// ---------------------------------------------------------------------------
// foldv: vctx[b,n] = mbar[b, n/64, :] @ Wv[:, n] + bv[n]
// (V-projection applied to the attention-weighted memory row). grid=(D/256, B).
// ---------------------------------------------------------------------------
__global__ void foldv_kernel(const float* __restrict__ mbar, // [B,16,1024]
                             const float* __restrict__ Wv,   // [1024,1024]
                             const float* __restrict__ bv,   // [1024]
                             float* __restrict__ vctx) {     // [B,1024]
  int b = blockIdx.y;
  int n = blockIdx.x * 256 + threadIdx.x;
  const float* mb = mbar + ((size_t)b * NHEAD + (n >> 6)) * DMODEL;
  float acc = bv[n];
  for (int k = 0; k < DMODEL; ++k) acc = fmaf(mb[k], Wv[(size_t)k * DMODEL + n], acc);
  vctx[b * DMODEL + n] = acc;
}

// ---------------------------------------------------------------------------
// layernorm over last dim (1024). grid = B, block = 256. Optional second copy.
// ---------------------------------------------------------------------------
__global__ void layernorm_kernel(const float* __restrict__ x,
                                 const float* __restrict__ g,
                                 const float* __restrict__ beta,
                                 float* __restrict__ y,
                                 float* __restrict__ ycopy) {
  int b = blockIdx.x;
  const float* xb = x + b * DMODEL;
  __shared__ float red[256];
  float s = 0.0f;
  for (int i = threadIdx.x; i < DMODEL; i += 256) s += xb[i];
  red[threadIdx.x] = s; __syncthreads();
  for (int off = 128; off > 0; off >>= 1) {
    if (threadIdx.x < off) red[threadIdx.x] += red[threadIdx.x + off];
    __syncthreads();
  }
  float mean = red[0] * (1.0f / DMODEL); __syncthreads();
  float vs = 0.0f;
  for (int i = threadIdx.x; i < DMODEL; i += 256) {
    float d = xb[i] - mean; vs += d * d;
  }
  red[threadIdx.x] = vs; __syncthreads();
  for (int off = 128; off > 0; off >>= 1) {
    if (threadIdx.x < off) red[threadIdx.x] += red[threadIdx.x + off];
    __syncthreads();
  }
  float rstd = rsqrtf(red[0] * (1.0f / DMODEL) + 1e-6f);
  for (int i = threadIdx.x; i < DMODEL; i += 256) {
    float v = (xb[i] - mean) * rstd * g[i] + beta[i];
    y[b * DMODEL + i] = v;
    if (ycopy) ycopy[b * DMODEL + i] = v;
  }
}

// ---------------------------------------------------------------------------
// gate: g = sigmoid([xa;xb] . fW + fb); out = g*xa + (1-g)*xb (+ resid)
// grid = B, block = 256.
// ---------------------------------------------------------------------------
__global__ void gate_kernel(const float* __restrict__ xa,
                            const float* __restrict__ xb,
                            const float* __restrict__ fW,   // [2048]
                            const float* __restrict__ fb,   // [1]
                            const float* __restrict__ resid,
                            float* __restrict__ out) {
  int b = blockIdx.x;
  const float* a = xa + b * DMODEL;
  const float* c = xb + b * DMODEL;
  __shared__ float red[256];
  float s = 0.0f;
  for (int i = threadIdx.x; i < DMODEL; i += 256)
    s += a[i] * fW[i] + c[i] * fW[DMODEL + i];
  red[threadIdx.x] = s; __syncthreads();
  for (int off = 128; off > 0; off >>= 1) {
    if (threadIdx.x < off) red[threadIdx.x] += red[threadIdx.x + off];
    __syncthreads();
  }
  float gate = 1.0f / (1.0f + __expf(-(red[0] + fb[0])));
  for (int i = threadIdx.x; i < DMODEL; i += 256) {
    float v = gate * a[i] + (1.0f - gate) * c[i];
    if (resid) v += resid[b * DMODEL + i];
    out[b * DMODEL + i] = v;
  }
}

// ---------------------------------------------------------------------------
// Host launcher
// ---------------------------------------------------------------------------
extern "C" void kernel_launch(void* const* d_in, const int* in_sizes, int n_in,
                              void* d_out, int out_size, void* d_ws, size_t ws_size,
                              hipStream_t stream) {
  (void)in_sizes; (void)n_in; (void)out_size; (void)ws_size;
  const int S = 2048, E = 256, P = 512;

  // setup_inputs() dict order, params dict flattened in insertion order.
  enum {
    IN_INPUTS = 0, IN_MEM, IN_SRC_MASK, IN_TGT_MASK, IN_ENT, IN_ENT_MASK,
    IN_ENT_SCORE, IN_P1, IN_P1_MASK,
    SELF_WQ, SELF_BQ, SELF_WK, SELF_BK, SELF_WV, SELF_BV, SELF_WO, SELF_BO,
    WORD_WQ, WORD_BQ, WORD_WK, WORD_BK, WORD_WV, WORD_BV, WORD_WO, WORD_BO,
    GRAPH_WQ, GRAPH_BQ, GRAPH_WK, GRAPH_BK, GRAPH_WV, GRAPH_BV, GRAPH_WO, GRAPH_BO,
    P1_WQ, P1_BQ, P1_WK, P1_BK, P1_WV, P1_BV, P1_WO, P1_BO,
    LN1_G, LN1_B, LN2_G, LN2_B, FFN_LN_G, FFN_LN_B,
    W1, B1, W2, B2, F1_W, F1_B, F2_W, F2_B
  };
  auto F = [&](int i) { return (const float*)d_in[i]; };
  auto U8 = [&](int i) { return (const unsigned char*)d_in[i]; };

  float* out0      = (float*)d_out;            // ff + h         [32,1,1024]
  float* all_input = (float*)d_out + 32768;    // input_norm     [32,1,1024]

  // workspace layout (floats)
  float* w = (float*)d_ws;
  float* input_norm = w;                 // 32768
  float* queryv     = w + 32768;         // 32768
  float* qn         = w + 65536;         // 32768
  float* qproj      = w + 98304;         // 3 x 32768
  float* vtmp       = w + 196608;        // 32768
  float* u_all      = w + 229376;        // 3 x 524288
  float* cb_all     = w + 1802240;       // 3 x 512
  float* sc_word    = w + 1803776;       // 32*16*2048 = 1048576
  float* sc_graph   = w + 2852352;       // 32*16*256  = 131072
  float* sc_p1      = w + 2983424;       // 32*16*512  = 262144
  float* mbar       = w + 3245568;       // 3 x 524288
  float* vctx       = w + 4818432;       // 3 x 32768
  float* xctx       = w + 4916736;       // 3 x 32768 (word, graph, p1)
  float* outctx     = w + 5015040;       // 32768
  float* hbuf       = w + 5047808;       // 32768
  float* inter      = w + 5080576;       // 32768
  float* ff1        = w + 5113344;       // 131072

  dim3 blk(256);

  // 1) input_norm = LN1(inputs); also emit all_input output.
  layernorm_kernel<<<NBATCH, blk, 0, stream>>>(F(IN_INPUTS), F(LN1_G), F(LN1_B),
                                               input_norm, all_input);
  // 2) self-attn (T=1 => softmax==1): query = Wo(Wv(input_norm)+bv)+bo + inputs
  gemm32_kernel<<<8, blk, 0, stream>>>(input_norm, F(SELF_WV), F(SELF_BV), nullptr,
                                       vtmp, DMODEL, DMODEL, 0);
  gemm32_kernel<<<8, blk, 0, stream>>>(vtmp, F(SELF_WO), F(SELF_BO), F(IN_INPUTS),
                                       queryv, DMODEL, DMODEL, 0);
  // 3) qn = LN2(query)
  layernorm_kernel<<<NBATCH, blk, 0, stream>>>(queryv, F(LN2_G), F(LN2_B), qn, nullptr);

  // per-source tables
  const float* mems[3]   = { F(IN_MEM), F(IN_ENT), F(IN_P1) };
  const int    slen[3]   = { S, E, P };
  const unsigned char* masks[3] = { U8(IN_SRC_MASK), U8(IN_ENT_MASK), U8(IN_P1_MASK) };
  const float* escores[3] = { nullptr, F(IN_ENT_SCORE), nullptr };
  const int wq[3] = { WORD_WQ, GRAPH_WQ, P1_WQ };
  const int bq[3] = { WORD_BQ, GRAPH_BQ, P1_BQ };
  const int wk[3] = { WORD_WK, GRAPH_WK, P1_WK };
  const int bk[3] = { WORD_BK, GRAPH_BK, P1_BK };
  const int wv[3] = { WORD_WV, GRAPH_WV, P1_WV };
  const int bv[3] = { WORD_BV, GRAPH_BV, P1_BV };
  const int wo[3] = { WORD_WO, GRAPH_WO, P1_WO };
  const int bo[3] = { WORD_BO, GRAPH_BO, P1_BO };
  float* scores[3] = { sc_word, sc_graph, sc_p1 };

  for (int x = 0; x < 3; ++x) {
    float* qx  = qproj + x * 32768;
    float* ux  = u_all + (size_t)x * 524288;
    float* cbx = cb_all + x * 512;
    float* mbx = mbar + (size_t)x * 524288;
    float* vcx = vctx + x * 32768;
    float* xcx = xctx + x * 32768;
    int Sx = slen[x];

    // 4) q = qn @ Wq + bq
    gemm32_kernel<<<8, blk, 0, stream>>>(qn, F(wq[x]), F(bq[x]), nullptr,
                                         qx, DMODEL, DMODEL, 0);
    // 5) fold K-projection into query: u, cbias
    attn_u_kernel<<<dim3(NHEAD, NBATCH), blk, 0, stream>>>(qx, F(wk[x]), F(bk[x]), ux, cbx);
    // 6) WMMA score GEMM: [Sx,1024] x [1024,16]
    attn_score_kernel<<<dim3(Sx / 128, NBATCH), blk, 0, stream>>>(mems[x], ux, scores[x], Sx);
    // 7) softmax (+ entity-score reweight for graph)
    softmax_kernel<<<dim3(NHEAD, NBATCH), blk, 0, stream>>>(scores[x], cbx, masks[x],
                                                            escores[x], Sx);
    // 8) probability-weighted memory rows (streaming, HBM-bound, async->LDS)
    attn_wsum_kernel<<<dim3(DMODEL / 256, NBATCH), blk, 0, stream>>>(mems[x], scores[x],
                                                                     mbx, Sx);
    // 9) apply folded V-projection, then output projection
    foldv_kernel<<<dim3(DMODEL / 256, NBATCH), blk, 0, stream>>>(mbx, F(wv[x]), F(bv[x]), vcx);
    gemm32_kernel<<<8, blk, 0, stream>>>(vcx, F(wo[x]), F(bo[x]), nullptr,
                                         xcx, DMODEL, DMODEL, 0);
  }

  // 10) gates
  gate_kernel<<<NBATCH, blk, 0, stream>>>(xctx, xctx + 32768, F(F1_W), F(F1_B),
                                          nullptr, outctx);                    // word/graph
  gate_kernel<<<NBATCH, blk, 0, stream>>>(outctx, xctx + 65536, F(F2_W), F(F2_B),
                                          queryv, hbuf);                       // (+query resid)

  // 11) FFN: LN -> W1 -> relu -> W2 -> residual
  layernorm_kernel<<<NBATCH, blk, 0, stream>>>(hbuf, F(FFN_LN_G), F(FFN_LN_B), inter, nullptr);
  gemm32_kernel<<<32, blk, 0, stream>>>(inter, F(W1), F(B1), nullptr,
                                        ff1, DMODEL, 4096, 1);
  gemm32_kernel<<<8, blk, 0, stream>>>(ff1, F(W2), F(B2), hbuf,
                                       out0, 4096, DMODEL, 0);
}
